// FrequencyDomainFilterLayer_80599356276759
// MI455X (gfx1250) — matrix operand order repaired and checked by reading, not compiled, not run on hardware
//
#include <hip/hip_runtime.h>

typedef _Float16 half_t;
typedef _Float16 v16h __attribute__((ext_vector_type(16)));
typedef float    v8f  __attribute__((ext_vector_type(8)));

#define BATCH 32
#define CIN   512
#define DM    1024

// ---------------------------------------------------------------------------
// K1: nw = softmax(relu(w_dyn)) along axis=1 for real & imag, store f16 [C,C]
// ---------------------------------------------------------------------------
__global__ __launch_bounds__(256) void k_nw(const float* __restrict__ wr,
                                            const float* __restrict__ wi,
                                            half_t* __restrict__ outr,
                                            half_t* __restrict__ outi) {
  int row = blockIdx.x;
  const float* w = (blockIdx.y == 0) ? wr : wi;
  half_t* out = (blockIdx.y == 0) ? outr : outi;
  __shared__ float red[256];
  int t = threadIdx.x;
  float v0 = fmaxf(w[row * CIN + t], 0.f);
  float v1 = fmaxf(w[row * CIN + t + 256], 0.f);
  float mx = fmaxf(v0, v1);
  red[t] = mx; __syncthreads();
  for (int off = 128; off > 0; off >>= 1) {
    if (t < off) red[t] = fmaxf(red[t], red[t + off]);
    __syncthreads();
  }
  mx = red[0]; __syncthreads();
  float e0 = expf(v0 - mx), e1 = expf(v1 - mx);
  red[t] = e0 + e1; __syncthreads();
  for (int off = 128; off > 0; off >>= 1) {
    if (t < off) red[t] += red[t + off];
    __syncthreads();
  }
  float inv = 1.f / red[0];
  out[row * CIN + t]       = (half_t)(e0 * inv);
  out[row * CIN + t + 256] = (half_t)(e1 * inv);
}

// ---------------------------------------------------------------------------
// K2: S = amp_st * filters * softmax(relu(w_st)) along D, f32 [C,D] (r & i);
//     also mf = softmax(mixing_factor, axis=1) [C,3]
// ---------------------------------------------------------------------------
__global__ __launch_bounds__(256) void k_static(const float* __restrict__ wsr,
                                                const float* __restrict__ wsi,
                                                const float* __restrict__ filters,
                                                const float* __restrict__ amp_st,
                                                const float* __restrict__ mixing,
                                                float* __restrict__ Sr,
                                                float* __restrict__ Si,
                                                float* __restrict__ mf) {
  int row = blockIdx.x;
  const float* w = (blockIdx.y == 0) ? wsr : wsi;
  float* S = (blockIdx.y == 0) ? Sr : Si;
  __shared__ float red[256];
  int t = threadIdx.x;
  float v[4], e[4];
  float mx = 0.f;
  #pragma unroll
  for (int q = 0; q < 4; ++q) {
    v[q] = fmaxf(w[(size_t)row * DM + t + q * 256], 0.f);
    mx = fmaxf(mx, v[q]);
  }
  red[t] = mx; __syncthreads();
  for (int off = 128; off > 0; off >>= 1) {
    if (t < off) red[t] = fmaxf(red[t], red[t + off]);
    __syncthreads();
  }
  mx = red[0]; __syncthreads();
  float sm = 0.f;
  #pragma unroll
  for (int q = 0; q < 4; ++q) { e[q] = expf(v[q] - mx); sm += e[q]; }
  red[t] = sm; __syncthreads();
  for (int off = 128; off > 0; off >>= 1) {
    if (t < off) red[t] += red[t + off];
    __syncthreads();
  }
  float inv = 1.f / red[0];
  #pragma unroll
  for (int q = 0; q < 4; ++q) {
    size_t o = (size_t)row * DM + t + q * 256;
    S[o] = e[q] * inv * filters[o] * amp_st[o];
  }
  if (blockIdx.y == 0 && t == 0) {
    float a = mixing[row * 3 + 0], b = mixing[row * 3 + 1], c = mixing[row * 3 + 2];
    float m = fmaxf(a, fmaxf(b, c));
    float ea = expf(a - m), eb = expf(b - m), ec = expf(c - m);
    float s = ea + eb + ec;
    mf[row * 3 + 0] = ea / s;
    mf[row * 3 + 1] = eb / s;
    mf[row * 3 + 2] = ec / s;
  }
}

// ---------------------------------------------------------------------------
// K3: per (b,c) row: thr = quantile_0.9(|x|) via bitonic sort (exact, linear
//     interp between order stats 920/921); write masked x as f16.
// ---------------------------------------------------------------------------
__global__ __launch_bounds__(256) void k_mask(const float* __restrict__ xr,
                                              const float* __restrict__ xi,
                                              half_t* __restrict__ mr,
                                              half_t* __restrict__ mi) {
  int row = blockIdx.x;                 // b*CIN + c
  size_t base = (size_t)row * DM;
  __shared__ float s[DM];
  int t = threadIdx.x;
  float rxr[4], rxi[4], rmg[4];
  #pragma unroll
  for (int q = 0; q < 4; ++q) {
    int d = t + q * 256;
    rxr[q] = xr[base + d];
    rxi[q] = xi[base + d];
    rmg[q] = sqrtf(rxr[q] * rxr[q] + rxi[q] * rxi[q]);
    s[d] = rmg[q];
  }
  __syncthreads();
  // bitonic sort ascending, 1024 elements, 512 compare-exchanges / stage
  for (int k = 2; k <= DM; k <<= 1) {
    for (int j = k >> 1; j > 0; j >>= 1) {
      for (int p = t; p < DM / 2; p += 256) {
        int i = ((p & ~(j - 1)) << 1) | (p & (j - 1));
        int ixj = i | j;
        float a = s[i], b = s[ixj];
        bool up = ((i & k) == 0);
        if ((a > b) == up) { s[i] = b; s[ixj] = a; }
      }
      __syncthreads();
    }
  }
  float pos = 0.9f * (float)(DM - 1);     // 920.7
  int lo = (int)pos;
  float fr = pos - (float)lo;
  float thr = s[lo] * (1.f - fr) + s[lo + 1] * fr;
  #pragma unroll
  for (int q = 0; q < 4; ++q) {
    int d = t + q * 256;
    bool m = rmg[q] > thr;
    mr[base + d] = (half_t)(m ? rxr[q] : 0.f);
    mi[base + d] = (half_t)(m ? rxi[q] : 0.f);
  }
}

// ---------------------------------------------------------------------------
// K4: fused complex GEMM (ws = nw @ conj(masked)) + full epilogue.
//     Block tile 64(M) x 128(N), 8 waves, wave tile 16(M) x 64(N).
//     Double-buffered LDS (one barrier/step); B tile staged transposed with
//     packed ds_store_b64 column-quads (bank-friendly: 20*n mod 64 permutes).
//     accR  = nr@mr + ni@mi   (chained into one accumulator)
//     accI1 = ni@mr ; accI2 = nr@mi ; ws_i = accI1 - accI2
// ---------------------------------------------------------------------------
#define PAD 40   // halves per LDS row (32 + pad); multiple of 8 -> 16B-aligned reads

__global__ __launch_bounds__(256) void k_gemm(
    const half_t* __restrict__ nwr, const half_t* __restrict__ nwi,
    const half_t* __restrict__ mr,  const half_t* __restrict__ mi,
    const float*  __restrict__ xr,  const float*  __restrict__ xi,
    const float*  __restrict__ amp_dyn,
    const float*  __restrict__ mix_r, const float* __restrict__ mix_i,
    const float*  __restrict__ Sr,  const float*  __restrict__ Si,
    const float*  __restrict__ mf,
    float* __restrict__ out) {
  __shared__ half_t sBr[2][128 * PAD];
  __shared__ half_t sBi[2][128 * PAD];

  int b = blockIdx.z;
  int nblock = blockIdx.x * 128;
  int mblock = blockIdx.y * 64;
  int t = threadIdx.x;
  int wave = t >> 5, lane = t & 31;
  int wm = wave & 3, wn = wave >> 2;
  int l15 = lane & 15, hf = lane >> 4;
  int mwave = mblock + wm * 16;

  // staging role: threads 0..127 -> real matrix, 128..255 -> imag matrix
  const half_t* stg_src = (t < 128) ? mr : mi;
  int w7 = t & 127;
  int s_no = w7 & 15;          // n-oct index: n = s_no*8 .. +7
  int s_kq = w7 >> 4;          // k-quad index: k = s_kq*4 .. +3

  v8f accR[4], accI1[4], accI2[4];
  #pragma unroll
  for (int s2 = 0; s2 < 4; ++s2)
    #pragma unroll
    for (int e = 0; e < 8; ++e) { accR[s2][e] = 0.f; accI1[s2][e] = 0.f; accI2[s2][e] = 0.f; }

  auto stage_load = [&](int step, uint4* v) {
    size_t g = ((size_t)(b * CIN + step * 32 + s_kq * 4)) * DM + nblock + s_no * 8;
    #pragma unroll
    for (int r = 0; r < 4; ++r) v[r] = *(const uint4*)(stg_src + g + (size_t)r * DM);
  };
  auto stage_store = [&](int buf, const uint4* v) {
    half_t* dst = (t < 128) ? sBr[buf] : sBi[buf];
    const unsigned short* p0 = (const unsigned short*)&v[0];
    const unsigned short* p1 = (const unsigned short*)&v[1];
    const unsigned short* p2 = (const unsigned short*)&v[2];
    const unsigned short* p3 = (const unsigned short*)&v[3];
    #pragma unroll
    for (int j = 0; j < 8; ++j) {          // column (n, k..k+3) -> one b64 store
      uint2 col;
      col.x = (unsigned)p0[j] | ((unsigned)p1[j] << 16);
      col.y = (unsigned)p2[j] | ((unsigned)p3[j] << 16);
      *(uint2*)(dst + (s_no * 8 + j) * PAD + s_kq * 4) = col;
    }
  };

  {
    uint4 v0[4];
    stage_load(0, v0);
    stage_store(0, v0);
  }
  __syncthreads();

  for (int step = 0; step < CIN / 32; ++step) {
    int cur = step & 1;
    int k0 = step * 32;
    uint4 vnext[4];
    if (step < CIN / 32 - 1) stage_load(step + 1, vnext);   // prefetch next tile

    // A fragments: lane l<16 holds row m=l, K 0..15 contiguous; lane l+16 same row,
    // K 16..31  (two b128 loads each, L2-resident nw matrices)
    union AFrag { uint4 u[2]; v16h h; };
    AFrag ar, ai;
    {
      size_t arow = (size_t)(mwave + l15) * CIN + k0 + hf * 16;
      ar.u[0] = *(const uint4*)(nwr + arow);
      ar.u[1] = *(const uint4*)(nwr + arow + 8);
      ai.u[0] = *(const uint4*)(nwi + arow);
      ai.u[1] = *(const uint4*)(nwi + arow + 8);
    }
    #pragma unroll
    for (int s2 = 0; s2 < 4; ++s2) {
      // B fragment: lane l<16 = column n, K 0..15 contiguous; lane l+16 K 16..31
      int nl = wn * 64 + s2 * 16 + l15;
      const uint4* bpr = (const uint4*)(&sBr[cur][nl * PAD + hf * 16]);
      const uint4* bpi = (const uint4*)(&sBi[cur][nl * PAD + hf * 16]);
      AFrag br, bi;
      br.u[0] = bpr[0]; br.u[1] = bpr[1];
      bi.u[0] = bpi[0]; bi.u[1] = bpi[1];
      accR[s2]  = __builtin_amdgcn_wmma_f32_16x16x32_f16(false, ar.h, false, br.h, (short)0, accR[s2],  false, false);
      accR[s2]  = __builtin_amdgcn_wmma_f32_16x16x32_f16(false, ai.h, false, bi.h, (short)0, accR[s2],  false, false);
      accI1[s2] = __builtin_amdgcn_wmma_f32_16x16x32_f16(false, ai.h, false, br.h, (short)0, accI1[s2], false, false);
      accI2[s2] = __builtin_amdgcn_wmma_f32_16x16x32_f16(false, ar.h, false, bi.h, (short)0, accI2[s2], false, false);
    }

    if (step < CIN / 32 - 1) stage_store(cur ^ 1, vnext);   // fill other buffer
    __syncthreads();
  }

  // fused epilogue: dynamic filter mix + static filter + outer softmax mixing
  #pragma unroll
  for (int v = 0; v < 8; ++v) {
    int m = mwave + v + hf * 8;           // C/D layout: lanes16-31 are M+8
    float mxr = mix_r[m], mxi = mix_i[m];
    float f0 = mf[m * 3 + 0], f1 = mf[m * 3 + 1], f2 = mf[m * 3 + 2];
    #pragma unroll
    for (int s2 = 0; s2 < 4; ++s2) {
      int n = nblock + wn * 64 + s2 * 16 + l15;
      size_t idx = ((size_t)(b * CIN + m)) * DM + n;
      float wsr = accR[s2][v];
      float wsi = accI1[s2][v] - accI2[s2][v];
      float xrv = xr[idx], xiv = xi[idx];
      float amp = amp_dyn[(size_t)m * DM + n];
      float adr = xrv * amp, adi = xiv * amp;
      float fr = adr * wsr - adi * wsi;          // filtered = adjusted * ws
      float fi = adr * wsi + adi * wsr;
      float dr = (1.f - mxr) * xrv + mxi * xiv + mxr * fr - mxi * fi;
      float di = (1.f - mxr) * xiv - mxi * xrv + mxr * fi + mxi * fr;
      float sr = Sr[(size_t)m * DM + n], si = Si[(size_t)m * DM + n];
      float str = xrv * sr - xiv * si;           // st_out = x * S
      float sti = xrv * si + xiv * sr;
      float2 o;
      o.x = xrv * f0 + str * f1 + dr * f2;
      o.y = xiv * f0 + sti * f1 + di * f2;
      *(float2*)(out + idx * 2) = o;
    }
  }
}

// ---------------------------------------------------------------------------
extern "C" void kernel_launch(void* const* d_in, const int* in_sizes, int n_in,
                              void* d_out, int out_size, void* d_ws, size_t ws_size,
                              hipStream_t stream) {
  (void)in_sizes; (void)n_in; (void)out_size; (void)ws_size;
  const float* x_r     = (const float*)d_in[0];
  const float* x_i     = (const float*)d_in[1];
  const float* amp_dyn = (const float*)d_in[2];
  const float* w_dyn_r = (const float*)d_in[3];
  const float* w_dyn_i = (const float*)d_in[4];
  const float* mix_r   = (const float*)d_in[5];
  const float* mix_i   = (const float*)d_in[6];
  const float* amp_st  = (const float*)d_in[7];
  const float* w_st_r  = (const float*)d_in[8];
  const float* w_st_i  = (const float*)d_in[9];
  const float* filters = (const float*)d_in[10];
  const float* mixing  = (const float*)d_in[11];
  float* out = (float*)d_out;

  char* ws = (char*)d_ws;
  const size_t MB = 1024 * 1024;
  half_t* nwr = (half_t*)(ws + 0);                    // 0.5 MB
  half_t* nwi = (half_t*)(ws + (size_t)CIN * CIN * 2);// 0.5 MB (ends at 1MB)
  float*  Sr  = (float*)(ws + 1 * MB);                // 2 MB
  float*  Si  = (float*)(ws + 3 * MB);                // 2 MB
  float*  mf  = (float*)(ws + 5 * MB);                // 6 KB
  half_t* mrp = (half_t*)(ws + 6 * MB);               // 32 MB
  half_t* mip = (half_t*)(ws + 40 * MB);              // 32 MB (ends at 72MB)

  k_nw<<<dim3(CIN, 2), 256, 0, stream>>>(w_dyn_r, w_dyn_i, nwr, nwi);
  k_static<<<dim3(CIN, 2), 256, 0, stream>>>(w_st_r, w_st_i, filters, amp_st,
                                             mixing, Sr, Si, mf);
  k_mask<<<dim3(BATCH * CIN), 256, 0, stream>>>(x_r, x_i, mrp, mip);
  k_gemm<<<dim3(DM / 128, CIN / 64, BATCH), 256, 0, stream>>>(
      nwr, nwi, mrp, mip, x_r, x_i, amp_dyn, mix_r, mix_i, Sr, Si, mf, out);
}
